// NTM_Associative_Recall_Model_35089882808661
// MI455X (gfx1250) — matrix operand order repaired and checked by reading, not compiled, not run on hardware
//
#include <hip/hip_runtime.h>
#include <hip/hip_bf16.h>
#include <math.h>

typedef _Float16 f16;
typedef __attribute__((ext_vector_type(16))) _Float16 v16h;
typedef __attribute__((ext_vector_type(8)))  _Float16 v8h;
typedef __attribute__((ext_vector_type(8)))  float    v8f;

#define BB 64          // batch
#define TT 128         // read timesteps
#define LL 32          // write timesteps
#define DD 256         // io dim
#define RR 1024        // rnn size
#define NN 512         // memory rows
#define MM 128         // memory cols
#define KCAT 1408      // D + M + R  (acat = [x | reads | h])
#define HEADD 524      // head params
#define HEADP 528      // head params padded to 16
#define KOUT 1152      // M + R
#define EPSF 1e-8f

union V16U { v16h v; v8h h[2]; };

__device__ __forceinline__ float sigmoidf_(float x) { return 1.0f / (1.0f + __expf(-x)); }
__device__ __forceinline__ float softplusf_(float x) { return (x > 20.f) ? x : log1pf(__expf(x)); }

// ---------------------------------------------------------------------------
// Generic WMMA GEMM: C[64 x nValid] = A_f16[64 x K] (row-major, stride lda)
//                                   @ Bfrag (pre-swizzled f16) + bias
// One wave -> one 16x16 output tile. 4 waves/block cover the 64 batch rows.
// grid.x = Npad/16 (one block per n-tile).
// ---------------------------------------------------------------------------
__global__ __launch_bounds__(128) void k_gemm(
    const f16* __restrict__ A, int lda, int K,
    const f16* __restrict__ Bfrag,
    const float* __restrict__ bias, int nValid,
    float* __restrict__ C, int ldc, int sigm)
{
  const int lane = threadIdx.x & 31;
  const int wave = threadIdx.x >> 5;
  const int nT = blockIdx.x;
  const int m0 = wave * 16;
  const int kTiles = K >> 5;

  // A: row m0 + lane%16; lane<16 -> K {0..15}, lane>=16 -> K {16..31}
  const int row = m0 + (lane & 15);
  const int kshift = (lane >> 4) << 4;             // 0 or 16
  const f16* abase = A + (size_t)row * lda + kshift;
  // B fragment: 512 halves per (nTile,kTile); lane's 16 halves contiguous
  const f16* bbase = Bfrag + (size_t)nT * kTiles * 512 + lane * 16;

  v8f acc = {0.f, 0.f, 0.f, 0.f, 0.f, 0.f, 0.f, 0.f};
#pragma unroll 2
  for (int kt = 0; kt < kTiles; ++kt) {
    V16U a, b;
    a.h[0] = *(const v8h*)(abase + kt * 32);
    a.h[1] = *(const v8h*)(abase + kt * 32 + 8);
    b.h[0] = *(const v8h*)(bbase + (size_t)kt * 512);
    b.h[1] = *(const v8h*)(bbase + (size_t)kt * 512 + 8);
    acc = __builtin_amdgcn_wmma_f32_16x16x32_f16(
        false, a.v, false, b.v, (short)0, acc, false, false);
  }

  const int nn = nT * 16 + (lane & 15);
  if (nn < nValid) {
    const float bv = bias ? bias[nn] : 0.f;
    const int mofs = (lane >> 4) << 3;             // 0 or 8
#pragma unroll
    for (int r = 0; r < 8; ++r) {
      float v = acc[r] + bv;
      if (sigm) v = 1.0f / (1.0f + __expf(-v));
      C[(size_t)(m0 + r + mofs) * ldc + nn] = v;
    }
  }
}

// ---------------------------------------------------------------------------
// Weight prep: convert fp32 -> f16 and swizzle into WMMA B-fragment order.
// Frag index: ((nT*kTiles + kT)*32 + lane)*16 + hh
//   n = nT*16 + lane%16 ;  k = kT*32 + (lane<16 ? 0 : 16) + hh
// ---------------------------------------------------------------------------
__global__ void k_prep_wz(const float* __restrict__ Wl, const float* __restrict__ Ul,
                          f16* __restrict__ dst)
{
  size_t idx = (size_t)blockIdx.x * 256 + threadIdx.x;
  const size_t total = (size_t)KCAT * 4096;
  if (idx >= total) return;
  int hh = idx & 15;
  int lane = (idx >> 4) & 31;
  size_t ft = idx >> 9;
  const int kTiles = KCAT / 32;
  int kT = (int)(ft % kTiles), nT = (int)(ft / kTiles);
  int n = nT * 16 + (lane & 15);
  int k = kT * 32 + ((lane >> 4) << 4) + hh;
  float v = (k < DD + MM) ? Wl[(size_t)k * 4096 + n] : Ul[(size_t)(k - (DD + MM)) * 4096 + n];
  dst[idx] = (f16)v;
}

__global__ void k_prep_wh(const float* __restrict__ Wh, f16* __restrict__ dst)
{
  size_t idx = (size_t)blockIdx.x * 256 + threadIdx.x;
  const size_t total = (size_t)RR * HEADP;
  if (idx >= total) return;
  int hh = idx & 15;
  int lane = (idx >> 4) & 31;
  size_t ft = idx >> 9;
  const int kTiles = RR / 32;
  int kT = (int)(ft % kTiles), nT = (int)(ft / kTiles);
  int n = nT * 16 + (lane & 15);
  int k = kT * 32 + ((lane >> 4) << 4) + hh;
  float v = (n < HEADD) ? Wh[(size_t)k * HEADD + n] : 0.f;
  dst[idx] = (f16)v;
}

// W_out rows permuted to match acat[256:1408) = [reads(128) | h(1024)]
__global__ void k_prep_wo(const float* __restrict__ Wo, f16* __restrict__ dst)
{
  size_t idx = (size_t)blockIdx.x * 256 + threadIdx.x;
  const size_t total = (size_t)KOUT * DD;
  if (idx >= total) return;
  int hh = idx & 15;
  int lane = (idx >> 4) & 31;
  size_t ft = idx >> 9;
  const int kTiles = KOUT / 32;
  int kT = (int)(ft % kTiles), nT = (int)(ft / kTiles);
  int n = nT * 16 + (lane & 15);
  int k = kT * 32 + ((lane >> 4) << 4) + hh;
  int srcrow = (k < MM) ? (RR + k) : (k - MM);   // reads rows first, then h rows
  dst[idx] = (f16)Wo[(size_t)srcrow * DD + n];
}

// ---------------------------------------------------------------------------
// State init: Mem = 1e-6, c = 0, acat = 0, w_r = w_w = 1/N
// ---------------------------------------------------------------------------
__global__ void k_init(float* __restrict__ Mem, float* __restrict__ c,
                       f16* __restrict__ acat, float* __restrict__ wr,
                       float* __restrict__ ww)
{
  size_t idx = (size_t)blockIdx.x * 256 + threadIdx.x;
  if (idx < (size_t)BB * NN * MM) Mem[idx] = 1e-6f;
  if (idx < (size_t)BB * RR) c[idx] = 0.f;
  if (idx < (size_t)BB * KCAT) acat[idx] = (f16)0.f;
  if (idx < (size_t)BB * NN) { wr[idx] = 1.0f / NN; ww[idx] = 1.0f / NN; }
}

// x_t (or zeros in the write phase) -> acat[:, 0:256] as f16
__global__ void k_xcat(const float* __restrict__ x, int t, f16* __restrict__ acat)
{
  int b = blockIdx.x, j = threadIdx.x;
  float v = (t < TT) ? x[((size_t)b * TT + t) * DD + j] : 0.f;
  acat[(size_t)b * KCAT + j] = (f16)v;
}

// LSTM gate fusion: z[64,4096] -> c, h ; h stored as f16 into acat[:,384:1408)
__global__ void k_lstm(const float* __restrict__ z, float* __restrict__ c,
                       f16* __restrict__ acat)
{
  int idx = blockIdx.x * 256 + threadIdx.x;     // 65536
  int b = idx >> 10, j = idx & 1023;
  const float* zr = z + (size_t)b * 4096;
  float zi = zr[j], zf = zr[1024 + j], zg = zr[2048 + j], zo = zr[3072 + j];
  float cn = sigmoidf_(zf) * c[idx] + sigmoidf_(zi) * tanhf(zg);
  c[idx] = cn;
  float h = sigmoidf_(zo) * tanhf(cn);
  acat[(size_t)b * KCAT + DD + MM + j] = (f16)h;
}

// ---------------------------------------------------------------------------
// Fused NTM addressing: one block per batch element, 512 threads (16 waves).
// Read head: content sim -> softmax -> interpolate -> shift -> sharpen -> read.
// Write head: same weighting, then erase/add memory update.
// ---------------------------------------------------------------------------
__global__ __launch_bounds__(NN) void k_address(
    const float* __restrict__ p, float* __restrict__ Mem,
    float* __restrict__ wr, float* __restrict__ ww, f16* __restrict__ acat)
{
  __shared__ float s_norm[NN];
  __shared__ float s_k[MM];
  __shared__ float s_w[NN];
  __shared__ float s_red[NN];
  __shared__ float s_er[MM];
  __shared__ float s_ad[MM];
  __shared__ float s_scal[8];   // beta, g, s0, s1, s2, gamma, |k|

  const int b = blockIdx.x, tid = threadIdx.x;
  float* MemB = Mem + (size_t)b * NN * MM;
  const float* pb = p + (size_t)b * HEADD;

  { // row norms of pre-update memory (shared by both heads, as in reference)
    float acc = 0.f;
    const float* row = MemB + (size_t)tid * MM;
    for (int m = 0; m < MM; ++m) { float v = row[m]; acc += v * v; }
    s_norm[tid] = sqrtf(acc);
  }
  __syncthreads();

  for (int head = 0; head < 2; ++head) {
    const int o = head ? (MM + 6) : 0;
    float* wprev = head ? ww : wr;
    if (tid < MM) s_k[tid] = tanhf(pb[o + tid]);
    if (tid == 0) {
      s_scal[0] = softplusf_(pb[o + MM]);          // beta
      s_scal[1] = sigmoidf_(pb[o + MM + 1]);       // g
      float s0 = pb[o + MM + 2], s1 = pb[o + MM + 3], s2 = pb[o + MM + 4];
      float mx = fmaxf(s0, fmaxf(s1, s2));
      float e0 = __expf(s0 - mx), e1 = __expf(s1 - mx), e2 = __expf(s2 - mx);
      float inv = 1.f / (e0 + e1 + e2);
      s_scal[2] = e0 * inv; s_scal[3] = e1 * inv; s_scal[4] = e2 * inv;
      s_scal[5] = 1.0f + softplusf_(pb[o + MM + 5]);  // gamma
    }
    if (head && tid < MM) {
      s_er[tid] = sigmoidf_(pb[o + MM + 6 + tid]);
      s_ad[tid] = tanhf(pb[o + 2 * MM + 6 + tid]);
    }
    __syncthreads();
    if (tid == 0) {
      float acc = 0.f;
      for (int m = 0; m < MM; ++m) acc += s_k[m] * s_k[m];
      s_scal[6] = sqrtf(acc);                       // ||k||
    }
    __syncthreads();

    { // content logits: beta * cosine sim
      const float* row = MemB + (size_t)tid * MM;
      float dot = 0.f;
      for (int m = 0; m < MM; ++m) dot += row[m] * s_k[m];
      float sim = dot / (s_norm[tid] * s_scal[6] + EPSF);
      s_w[tid] = s_scal[0] * sim;
    }
    __syncthreads();

    // softmax over N=512
    s_red[tid] = s_w[tid];
    __syncthreads();
    for (int st = NN / 2; st > 0; st >>= 1) {
      if (tid < st) s_red[tid] = fmaxf(s_red[tid], s_red[tid + st]);
      __syncthreads();
    }
    float mx = s_red[0];
    __syncthreads();
    float e = __expf(s_w[tid] - mx);
    s_red[tid] = e;
    __syncthreads();
    for (int st = NN / 2; st > 0; st >>= 1) {
      if (tid < st) s_red[tid] += s_red[tid + st];
      __syncthreads();
    }
    float wc = e / s_red[0];

    // interpolate with previous weighting
    float g = s_scal[1];
    float wg = g * wc + (1.f - g) * wprev[(size_t)b * NN + tid];
    __syncthreads();
    s_w[tid] = wg;
    __syncthreads();

    // circular shift over {-1,0,+1}, then sharpen
    float wsft = s_scal[2] * s_w[(tid + 1) & (NN - 1)]
               + s_scal[3] * s_w[tid]
               + s_scal[4] * s_w[(tid + NN - 1) & (NN - 1)];
    float wp = __powf(wsft + EPSF, s_scal[5]);
    s_red[tid] = wp;
    __syncthreads();
    for (int st = NN / 2; st > 0; st >>= 1) {
      if (tid < st) s_red[tid] += s_red[tid + st];
      __syncthreads();
    }
    float w = wp / s_red[0];
    __syncthreads();
    s_w[tid] = w;
    wprev[(size_t)b * NN + tid] = w;
    __syncthreads();

    if (head == 0) {
      // reads[m] = sum_n w[n] * Mem[n][m]  (4 partial chunks of 128 rows)
      const int m = tid & (MM - 1);
      const int chunk = tid >> 7;
      float acc = 0.f;
      for (int n = chunk * 128; n < chunk * 128 + 128; ++n)
        acc += s_w[n] * MemB[(size_t)n * MM + m];
      s_red[tid] = acc;
      __syncthreads();
      if (tid < MM) {
        float r = s_red[tid] + s_red[tid + 128] + s_red[tid + 256] + s_red[tid + 384];
        acat[(size_t)b * KCAT + DD + tid] = (f16)r;
      }
      __syncthreads();
    } else {
      // Mem = Mem*(1 - w*erase) + w*add  (coalesced)
      for (int idx = tid; idx < NN * MM; idx += NN) {
        int n = idx >> 7, m = idx & (MM - 1);
        float wv = s_w[n];
        MemB[idx] = MemB[idx] * (1.f - wv * s_er[m]) + wv * s_ad[m];
      }
    }
  }
}

// ---------------------------------------------------------------------------
extern "C" void kernel_launch(void* const* d_in, const int* in_sizes, int n_in,
                              void* d_out, int out_size, void* d_ws, size_t ws_size,
                              hipStream_t stream)
{
  (void)in_sizes; (void)n_in; (void)out_size; (void)ws_size;
  const float* x  = (const float*)d_in[0];   // [64,128,256]
  const float* Wl = (const float*)d_in[1];   // [384,4096]
  const float* Ul = (const float*)d_in[2];   // [1024,4096]
  const float* bl = (const float*)d_in[3];   // [4096]
  const float* Wh = (const float*)d_in[4];   // [1024,524]
  const float* bh = (const float*)d_in[5];   // [524]
  const float* Wo = (const float*)d_in[6];   // [1152,256]
  const float* bo = (const float*)d_in[7];   // [256]
  float* out = (float*)d_out;                // [64,32,256]

  char* ws = (char*)d_ws;
  size_t off = 0;
  auto alloc = [&](size_t bytes) -> char* {
    char* r = ws + off;
    off = (off + bytes + 255) & ~(size_t)255;
    return r;
  };
  f16*  WzF  = (f16*)alloc((size_t)KCAT * 4096 * 2);   // 11.5 MB
  f16*  WhF  = (f16*)alloc((size_t)RR * HEADP * 2);    // 1.1 MB
  f16*  WoF  = (f16*)alloc((size_t)KOUT * DD * 2);     // 0.6 MB
  f16*  acat = (f16*)alloc((size_t)BB * KCAT * 2);     // [x|reads|h] f16
  float* z   = (float*)alloc((size_t)BB * 4096 * 4);
  float* pP  = (float*)alloc((size_t)BB * HEADD * 4);
  float* cS  = (float*)alloc((size_t)BB * RR * 4);
  float* wrS = (float*)alloc((size_t)BB * NN * 4);
  float* wwS = (float*)alloc((size_t)BB * NN * 4);
  float* Mem = (float*)alloc((size_t)BB * NN * MM * 4); // 16.8 MB

  // one-time (per call) weight swizzle + state init
  k_prep_wz<<<((size_t)KCAT * 4096 + 255) / 256, 256, 0, stream>>>(Wl, Ul, WzF);
  k_prep_wh<<<((size_t)RR * HEADP + 255) / 256, 256, 0, stream>>>(Wh, WhF);
  k_prep_wo<<<((size_t)KOUT * DD + 255) / 256, 256, 0, stream>>>(Wo, WoF);
  k_init<<<((size_t)BB * NN * MM + 255) / 256, 256, 0, stream>>>(Mem, cS, acat, wrS, wwS);

  for (int t = 0; t < TT + LL; ++t) {
    k_xcat<<<BB, DD, 0, stream>>>(x, t, acat);
    // z = acat @ [W_lstm;U_lstm] + b_lstm           (K=1408, N=4096)
    k_gemm<<<4096 / 16, 128, 0, stream>>>(acat, KCAT, KCAT, WzF, bl, 4096, z, 4096, 0);
    k_lstm<<<(BB * RR) / 256, 256, 0, stream>>>(z, cS, acat);
    // p = h @ W_heads + b_heads                     (K=1024, N=524)
    k_gemm<<<HEADP / 16, 128, 0, stream>>>(acat + DD + MM, KCAT, RR, WhF, bh, HEADD, pP, HEADD, 0);
    k_address<<<BB, NN, 0, stream>>>(pP, Mem, wrS, wwS, acat);
    if (t >= TT) {
      // out = sigmoid([reads,h] @ W_out' + b_out)   (K=1152, N=256), write phase only
      k_gemm<<<DD / 16, 128, 0, stream>>>(acat + DD, KCAT, KOUT, WoF, bo, DD,
                                          out + (size_t)(t - TT) * DD, LL * DD, 1);
    }
  }
}
